// SymmetricSupConLoss_43250320671200
// MI455X (gfx1250) — matrix operand order repaired and checked by reading, compiled, not run on hardware
//
#include <hip/hip_runtime.h>
#include <cmath>

// ---------------------------------------------------------------------------
// SymmetricSupConLoss for MI455X (gfx1250, wave32, WMMA).
//
// loss = ( sum_i cnt_i*lse_row_i + sum_j cnt_j*lse_col_j - 2*S ) / (2N)
//   lse = M0 + log(sum exp(logit - M0)),  M0 = 1/T (valid since |cos| <= 1)
//   S   = sum over positive (same-label) pairs of logit
// GEMM runs in bf16 WMMA (16x16x32) with f32 accumulation; softmax statistics
// are fused into the GEMM epilogue so the 64MB logits matrix is never stored.
// 256-thread blocks (8 waves, 4x2), 128x128 tile per block, 32x64 per wave
// (8 v_wmma per 32-wide K step).
// ---------------------------------------------------------------------------

typedef __attribute__((ext_vector_type(16))) __bf16 v16bf;
typedef __attribute__((ext_vector_type(8)))  float  v8f;
typedef unsigned short u16;

#define NROWS 4096
#define DDIM  768
#define NCLS  128
#define INV_T 14.285714285714286f   // 1/0.07

union Frag {
    uint4 q[2];
    v16bf v;
};

static __device__ inline u16 f2bf(float f) {
    unsigned u = __float_as_uint(f);
    unsigned r = u + 0x7FFFu + ((u >> 16) & 1u);   // round-to-nearest-even
    return (u16)(r >> 16);
}

static __device__ inline v8f wmma_bf16(const Frag& a, const Frag& b, v8f c) {
    return __builtin_amdgcn_wmma_f32_16x16x32_bf16(
        /*neg_a=*/false, a.v, /*neg_b=*/false, b.v,
        /*c_mod=*/(short)0, c, /*reuse_a=*/false, /*reuse_b=*/false);
}

// ------------------------------- kernels -----------------------------------

__global__ void init_zero_kernel(unsigned* p, int n) {
    int i = blockIdx.x * blockDim.x + threadIdx.x;
    if (i < n) p[i] = 0u;
}

__global__ void count_kernel(const int* __restrict__ labels, int* counts) {
    int i = blockIdx.x * blockDim.x + threadIdx.x;
    if (i < NROWS) atomicAdd(&counts[labels[i]], 1);
}

// One block (128 threads) per row; rows [0,4096) = img, [4096,8192) = txt.
__global__ __launch_bounds__(128)
void normalize_kernel(const float* __restrict__ img, const float* __restrict__ txt,
                      u16* __restrict__ imgN, u16* __restrict__ txtN) {
    int row = blockIdx.x;
    const float* src;
    u16* dst;
    if (row < NROWS) { src = img + (size_t)row * DDIM;           dst = imgN + (size_t)row * DDIM; }
    else             { src = txt + (size_t)(row - NROWS) * DDIM; dst = txtN + (size_t)(row - NROWS) * DDIM; }

    int t = threadIdx.x;                 // 0..127, DDIM/128 = 6 elems each
    float v[6];
    float ss = 0.0f;
#pragma unroll
    for (int e = 0; e < 6; ++e) { v[e] = src[t + 128 * e]; ss += v[e] * v[e]; }

#pragma unroll
    for (int m = 16; m >= 1; m >>= 1) ss += __shfl_xor(ss, m, 32);

    __shared__ float wsum[4];
    int lane = t & 31, wv = t >> 5;
    if (lane == 0) wsum[wv] = ss;
    __syncthreads();
    float tot   = wsum[0] + wsum[1] + wsum[2] + wsum[3];
    float scale = 1.0f / fmaxf(sqrtf(tot), 1e-8f);
#pragma unroll
    for (int e = 0; e < 6; ++e) dst[t + 128 * e] = f2bf(v[e] * scale);
}

// 128x128 logits tile per workgroup; 8 waves (4 row-blocks x 2 col-blocks),
// each wave owns 32 rows x 64 cols = 2x4 tiles of v_wmma_f32_16x16x32_bf16.
// Fused exp/row/col/mask epilogue.
__global__ __launch_bounds__(256)
void gemm_stats_kernel(const u16* __restrict__ imgN, const u16* __restrict__ txtN,
                       const int* __restrict__ labels,
                       float* __restrict__ rowSum, float* __restrict__ colSum,
                       float* __restrict__ S) {
    const int iBase = blockIdx.y * 128;
    const int jBase = blockIdx.x * 128;
    const int t    = threadIdx.x;
    const int lane = t & 31;
    const int wv   = t >> 5;
    const int wm   = wv >> 1;      // wave row (0..3) -> 32-row block
    const int wn   = wv & 1;       // wave col (0..1) -> 64-col block
    const int l15  = lane & 15;
    const bool hi  = lane >= 16;

    __shared__ int lblI[128], lblJ[128];
    if (t < 128) lblI[t] = labels[iBase + t];
    else         lblJ[t - 128] = labels[jBase + t - 128];
    __syncthreads();

    v8f zero = {0.f, 0.f, 0.f, 0.f, 0.f, 0.f, 0.f, 0.f};
    v8f acc[2][4];
#pragma unroll
    for (int mt = 0; mt < 2; ++mt)
#pragma unroll
        for (int nt = 0; nt < 4; ++nt) acc[mt][nt] = zero;

    // A fragment (16x32 bf16, ISA layout): lanes 0-15 hold K[0..7]+K[16..23],
    // lanes 16-31 hold K[8..15]+K[24..31] of row M = l15.
    const u16* aRow0 = imgN + (size_t)(iBase + wm * 32 + l15) * DDIM;
    const u16* aRow1 = aRow0 + (size_t)16 * DDIM;
    // B fragment (32x16 bf16): lane column N = l15; lanes 0-15 K[0..7]+K[16..23],
    // lanes 16-31 K[8..15]+K[24..31] (same K order as A).
    const u16* bRow[4];
#pragma unroll
    for (int nt = 0; nt < 4; ++nt)
        bRow[nt] = txtN + (size_t)(jBase + wn * 64 + nt * 16 + l15) * DDIM;
    const int aOff = hi ? 8 : 0;
    const int bOff = hi ? 8 : 0;

    for (int k0 = 0; k0 < DDIM; k0 += 32) {
        Frag a0, a1, b[4];
        const u16* pa0 = aRow0 + k0 + aOff;
        a0.q[0] = *(const uint4*)(pa0);
        a0.q[1] = *(const uint4*)(pa0 + 16);
        const u16* pa1 = aRow1 + k0 + aOff;
        a1.q[0] = *(const uint4*)(pa1);
        a1.q[1] = *(const uint4*)(pa1 + 16);
#pragma unroll
        for (int nt = 0; nt < 4; ++nt) {
            const u16* pb = bRow[nt] + k0 + bOff;
            b[nt].q[0] = *(const uint4*)(pb);
            b[nt].q[1] = *(const uint4*)(pb + 16);
        }
#pragma unroll
        for (int nt = 0; nt < 4; ++nt) acc[0][nt] = wmma_bf16(a0, b[nt], acc[0][nt]);
#pragma unroll
        for (int nt = 0; nt < 4; ++nt) acc[1][nt] = wmma_bf16(a1, b[nt], acc[1][nt]);
    }

    // Epilogue. C layout: VGPR r -> M = r (lanes 0-15) or r+8 (lanes 16-31),
    // N = l15.
    const int iHalf = hi ? 8 : 0;
    float rowPart[2][8];
#pragma unroll
    for (int mt = 0; mt < 2; ++mt)
#pragma unroll
        for (int r = 0; r < 8; ++r) rowPart[mt][r] = 0.0f;
    float colPart[4] = {0.0f, 0.0f, 0.0f, 0.0f};
    float sS = 0.0f;

#pragma unroll
    for (int mt = 0; mt < 2; ++mt) {
#pragma unroll
        for (int nt = 0; nt < 4; ++nt) {
#pragma unroll
            for (int r = 0; r < 8; ++r) {
                float dot   = acc[mt][nt][r];
                float logit = dot * INV_T;
                float e     = __expf(logit - INV_T);   // exp(logit - M0)
                rowPart[mt][r] += e;
                colPart[nt]    += e;
                int iL = wm * 32 + mt * 16 + r + iHalf;
                int jL = wn * 64 + nt * 16 + l15;
                if (lblI[iL] == lblJ[jL]) sS += logit;
            }
        }
    }

    // Column sums: lanes l and l+16 cover the two M halves of one column.
#pragma unroll
    for (int nt = 0; nt < 4; ++nt) {
        float c = colPart[nt] + __shfl_xor(colPart[nt], 16, 32);
        if (!hi) atomicAdd(&colSum[jBase + wn * 64 + nt * 16 + l15], c);
    }
    // Row sums: reduce across the 16 lanes of each half-wave.
#pragma unroll
    for (int mt = 0; mt < 2; ++mt) {
#pragma unroll
        for (int r = 0; r < 8; ++r) {
            float rs = rowPart[mt][r];
            rs += __shfl_xor(rs, 1, 32);
            rs += __shfl_xor(rs, 2, 32);
            rs += __shfl_xor(rs, 4, 32);
            rs += __shfl_xor(rs, 8, 32);
            if (l15 == r)
                atomicAdd(&rowSum[iBase + wm * 32 + mt * 16 + r + iHalf], rs);
        }
    }
    // Positive-pair logit sum.
#pragma unroll
    for (int m = 16; m >= 1; m >>= 1) sS += __shfl_xor(sS, m, 32);
    if (lane == 0) atomicAdd(S, sS);
}

__global__ __launch_bounds__(256)
void finalize_kernel(const float* __restrict__ rowSum, const float* __restrict__ colSum,
                     const int* __restrict__ labels, const int* __restrict__ counts,
                     const float* __restrict__ S, float* __restrict__ out) {
    float acc = 0.0f;
    for (int i = threadIdx.x; i < NROWS; i += 256) {
        float c = (float)counts[labels[i]];
        acc += c * (2.0f * INV_T + logf(rowSum[i]) + logf(colSum[i]));
    }
#pragma unroll
    for (int m = 16; m >= 1; m >>= 1) acc += __shfl_xor(acc, m, 32);
    __shared__ float wsum[8];
    int lane = threadIdx.x & 31, wv = threadIdx.x >> 5;
    if (lane == 0) wsum[wv] = acc;
    __syncthreads();
    if (threadIdx.x == 0) {
        float tot = 0.0f;
#pragma unroll
        for (int w = 0; w < 8; ++w) tot += wsum[w];
        out[0] = (tot - 2.0f * S[0]) / (2.0f * (float)NROWS);
    }
}

// ------------------------------ launcher -----------------------------------

extern "C" void kernel_launch(void* const* d_in, const int* in_sizes, int n_in,
                              void* d_out, int out_size, void* d_ws, size_t ws_size,
                              hipStream_t stream) {
    const float* img    = (const float*)d_in[0];
    const float* txt    = (const float*)d_in[1];
    const int*   labels = (const int*)d_in[2];
    float*       out    = (float*)d_out;

    char* ws = (char*)d_ws;
    const size_t embBytes = (size_t)NROWS * DDIM * sizeof(u16);  // 6 MB each
    u16*   imgN   = (u16*)ws;
    u16*   txtN   = (u16*)(ws + embBytes);
    float* rowSum = (float*)(ws + 2 * embBytes);
    float* colSum = rowSum + NROWS;
    int*   counts = (int*)(colSum + NROWS);
    float* S      = (float*)(counts + NCLS);

    // Zero stats region (rowSum | colSum | counts | S are contiguous).
    const int statWords = NROWS + NROWS + NCLS + 1;
    init_zero_kernel<<<(statWords + 511) / 512, 512, 0, stream>>>((unsigned*)rowSum, statWords);
    count_kernel<<<(NROWS + 255) / 256, 256, 0, stream>>>(labels, counts);
    normalize_kernel<<<2 * NROWS, 128, 0, stream>>>(img, txt, imgN, txtN);

    dim3 grid(NROWS / 128, NROWS / 128);   // 128x128 logits tile per workgroup
    gemm_stats_kernel<<<grid, 256, 0, stream>>>(imgN, txtN, labels, rowSum, colSum, S);

    finalize_kernel<<<1, 256, 0, stream>>>(rowSum, colSum, labels, counts, S, out);
}
